// BertTransformerWSD_5901285065183
// MI455X (gfx1250) — matrix operand order repaired and checked by reading, not compiled, not run on hardware
//
#include <hip/hip_runtime.h>
#include <hip/hip_bf16.h>

// ---------------- problem constants ----------------
#define B_      32
#define S_      256
#define T_      128
#define DBERT   768
#define DPOS    256
#define D_      1024
#define H_      16
#define DH_     64
#define FF_     4096
#define NSENSE_ 5000
#define NSENSEP 5120               // padded to a multiple of 128
#define M4_     (B_ * T_)          // 4096 token rows
#define SCALE_  32.0f              // sqrt(1024)
#define ASCALE_ 0.125f             // 1/sqrt(64)

typedef __bf16 bf16_t;
typedef __attribute__((ext_vector_type(4)))  __bf16 v4bf;
typedef __attribute__((ext_vector_type(8)))  __bf16 v8bf;
typedef __attribute__((ext_vector_type(16))) __bf16 v16bf;
typedef __attribute__((ext_vector_type(8)))  float  v8f;
typedef __attribute__((ext_vector_type(4)))  unsigned int v4u;
typedef __attribute__((ext_vector_type(4)))  int  v4i;
typedef __attribute__((ext_vector_type(8)))  int  v8i;

__device__ __forceinline__ v8f wmma_bf16(v16bf a, v16bf b, v8f c) {
  return __builtin_amdgcn_wmma_f32_16x16x32_bf16(false, a, false, b,
                                                 (short)0, c, false, false);
}

__device__ __forceinline__ v16bf merge16(v8bf lo, v8bf hi) {
  return __builtin_shufflevector(lo, hi, 0,1,2,3,4,5,6,7,8,9,10,11,12,13,14,15);
}

// -------- Tensor Data Mover: 2D bf16 tile (tile1 rows x tile0 elems) -> LDS ------
// D# layout per CDNA5 ISA.
// pad_int_code: row length in DWORDs = 2<<pad_int_code ; pad_amt_code: pad DWORDs-1.
__device__ __forceinline__ void tdm_load_2d(unsigned int lds_addr, const void* gsrc,
                                            unsigned int tile0, unsigned int tile1,
                                            unsigned long long stride0_elems,
                                            unsigned int pad_int_code,
                                            unsigned int pad_amt_code) {
  unsigned long long ga = (unsigned long long)(uintptr_t)gsrc;
  v4u g0;
  g0[0] = 1u;                                             // count=1 (valid user D#)
  g0[1] = lds_addr;                                       // LDS byte address
  g0[2] = (unsigned int)(ga & 0xFFFFFFFFu);               // global_addr[31:0]
  g0[3] = (unsigned int)((ga >> 32) & 0x01FFFFFFu) | (2u << 30); // addr[56:32]|type=2
  v8i g1;
  unsigned int w0 = (1u << 16)                            // data_size = 2 bytes
                  | (1u << 20)                            // pad_enable
                  | (pad_int_code << 22) | (pad_amt_code << 25);
  g1[0] = (int)w0;
  g1[1] = (int)((tile0 & 0xFFFFu) << 16);                 // tensor_dim0 = tile0
  g1[2] = (int)((tile0 >> 16) | ((tile1 & 0xFFFFu) << 16)); // tensor_dim1 = tile1
  g1[3] = (int)((tile1 >> 16) | (tile0 << 16));           // tile_dim0
  g1[4] = (int)(tile1 & 0xFFFFu);                         // tile_dim1 (tile_dim2=0)
  g1[5] = (int)(unsigned int)(stride0_elems & 0xFFFFFFFFu);
  g1[6] = (int)(unsigned int)((stride0_elems >> 32) & 0xFFFFu); // dim1_stride=0 (2D)
  g1[7] = 0;
  v4i z4 = {0, 0, 0, 0};
#if defined(__clang_major__) && (__clang_major__ >= 23)
  v8i z8 = {0, 0, 0, 0, 0, 0, 0, 0};
  __builtin_amdgcn_tensor_load_to_lds(g0, g1, z4, z4, z8, 0);
#else
  __builtin_amdgcn_tensor_load_to_lds(g0, g1, z4, z4, 0);
#endif
}

__device__ __forceinline__ unsigned int lds_off(const void* p) {
  return (unsigned int)(uintptr_t)p;   // flat LDS pointer: low 32 bits = LDS address
}

// ---------------- weight convert: f32 KxN -> bf16 transposed NdstxK --------------
__global__ __launch_bounds__(256)
void transpose_to_bf16_k(const float* __restrict__ src, bf16_t* __restrict__ dst,
                         int K, int N) {
  __shared__ float tile[32][33];
  const int kb = blockIdx.y * 32, nb = blockIdx.x * 32;
  const int tx = threadIdx.x & 31, ty = threadIdx.x >> 5;   // 32 x 8
  for (int i = 0; i < 32; i += 8) {
    const int k = kb + ty + i, n = nb + tx;
    tile[ty + i][tx] = (n < N) ? src[(size_t)k * N + n] : 0.0f;
  }
  __syncthreads();
  for (int i = 0; i < 32; i += 8) {
    const int n = nb + ty + i, k = kb + tx;
    dst[(size_t)n * K + k] = (bf16_t)tile[tx][ty + i];
  }
}

// ---------------- segment-mean pool + pos embed + *sqrt(D) ----------------
__global__ __launch_bounds__(256)
void pool_embed_k(const float* __restrict__ x, const int* __restrict__ word_ids,
                  const int* __restrict__ pos_tags, const float* __restrict__ pos_table,
                  float* __restrict__ hf, bf16_t* __restrict__ hb) {
  const int bt = blockIdx.x, b = bt / T_, t = bt % T_;
  const int tid = threadIdx.x;
  __shared__ int cnt2[2];
  if (tid < 2) cnt2[tid] = 0;
  __syncthreads();
  {
    int w = word_ids[b * S_ + tid];
    if (w <  t) atomicAdd(&cnt2[0], 1);
    if (w <= t) atomicAdd(&cnt2[1], 1);
  }
  __syncthreads();
  const int start = cnt2[0], end = cnt2[1], count = end - start;
  const float inv = (count > 0) ? (SCALE_ / (float)count) : 0.0f;
  for (int d = tid; d < DBERT; d += 256) {
    float s = 0.0f;
    for (int si = start; si < end; ++si)
      s += x[((size_t)b * S_ + si) * DBERT + d];
    float y = s * inv;
    hf[(size_t)bt * D_ + d] = y;
    hb[(size_t)bt * D_ + d] = (bf16_t)y;
  }
  const int tag = pos_tags[b * T_ + t];
  for (int d = tid; d < DPOS; d += 256) {
    float y = pos_table[tag * DPOS + d] * SCALE_;
    hf[(size_t)bt * D_ + DBERT + d] = y;
    hb[(size_t)bt * D_ + DBERT + d] = (bf16_t)y;
  }
}

// ------- WMMA GEMM: Y(M,Nout) = A(M,K)@Wt^T + bias, Wt stored N x K (bf16) -------
// Both tiles DMA'd by the Tensor Data Mover, double-buffered so the TDM for step
// s+1 overlaps the WMMA compute of step s. 256 thr = 8 waves (2x4), block tile
// 128x128, wave tile 64x32. EPI: 0=bf16, 1=relu->bf16, 2=f32 (compile-time).
template <int EPI>
__global__ __launch_bounds__(256)
void gemm_wmma_k(const bf16_t* __restrict__ A, const bf16_t* __restrict__ Wt,
                 const float* __restrict__ bias, void* __restrict__ out,
                 int Nout, int K) {
  __shared__ bf16_t As[2][128][48];   // [m][k]: 16-DW rows + 8-DW TDM pad -> stride 48
  __shared__ bf16_t Bs[2][128][48];   // [n][k]
  const int tid = threadIdx.x;
  const int lane = tid & 31, wid = tid >> 5;
  const int wm = wid >> 2, wn = wid & 3;
  const int hl = lane >> 4, r16 = lane & 15;
  const int tileM = blockIdx.y * 128, tileN = blockIdx.x * 128;
  const bf16_t* abase = A  + (size_t)tileM * K;
  const bf16_t* bbase = Wt + (size_t)tileN * K;

  v8f acc[4][2];
  for (int mt = 0; mt < 4; ++mt)
    for (int nt = 0; nt < 2; ++nt)
      for (int r = 0; r < 8; ++r) acc[mt][nt][r] = 0.0f;

  const int steps = K >> 5;
  if (wid == 0) {
    tdm_load_2d(lds_off(&As[0][0][0]), abase, 32u, 128u, (unsigned long long)K, 3u, 7u);
    tdm_load_2d(lds_off(&Bs[0][0][0]), bbase, 32u, 128u, (unsigned long long)K, 3u, 7u);
    __builtin_amdgcn_s_wait_tensorcnt(0);
  }
  __syncthreads();

  for (int s = 0; s < steps; ++s) {
    const int cur = s & 1;
    if ((s + 1 < steps) && (wid == 0)) {
      const int nxt = cur ^ 1;
      const bf16_t* an = abase + (size_t)(s + 1) * 32;
      const bf16_t* bn = bbase + (size_t)(s + 1) * 32;
      tdm_load_2d(lds_off(&As[nxt][0][0]), an, 32u, 128u, (unsigned long long)K, 3u, 7u);
      tdm_load_2d(lds_off(&Bs[nxt][0][0]), bn, 32u, 128u, (unsigned long long)K, 3u, 7u);
    }
    v16bf bfrag[2];
    for (int nt = 0; nt < 2; ++nt)
      bfrag[nt] = merge16(*(const v8bf*)&Bs[cur][wn * 32 + nt * 16 + r16][hl * 8],
                          *(const v8bf*)&Bs[cur][wn * 32 + nt * 16 + r16][16 + hl * 8]);
    for (int mt = 0; mt < 4; ++mt) {
      const int arow = wm * 64 + mt * 16 + r16;
      v8bf alo = *(const v8bf*)&As[cur][arow][hl * 8];
      v8bf ahi = *(const v8bf*)&As[cur][arow][16 + hl * 8];
      v16bf afrag = merge16(alo, ahi);
      for (int nt = 0; nt < 2; ++nt)
        acc[mt][nt] = wmma_bf16(afrag, bfrag[nt], acc[mt][nt]);
    }
    if (s + 1 < steps) {
      if (wid == 0) __builtin_amdgcn_s_wait_tensorcnt(0);
      __syncthreads();
    }
  }
  // epilogue: C layout lanes0-15: vgpr r -> M=r, N=lane; lanes16-31: M=8+r
  const int row0 = tileM + wm * 64 + hl * 8;
  for (int nt = 0; nt < 2; ++nt) {
    const int col = tileN + wn * 32 + nt * 16 + r16;
    if (col >= Nout) continue;
    const float bv = bias[col];
    size_t idx = (size_t)row0 * Nout + col;
    for (int mt = 0; mt < 4; ++mt) {
      size_t ridx = idx + (size_t)(mt * 16) * Nout;
      for (int r = 0; r < 8; ++r, ridx += Nout) {
        float v = acc[mt][nt][r] + bv;
        if (EPI == 1) v = fmaxf(v, 0.0f);
        if (EPI == 2) ((float*)out)[ridx] = v;
        else          ((bf16_t*)out)[ridx] = (bf16_t)v;
      }
    }
  }
}

// ---------------- attention scores: S = Q@K^T * scale + key mask ----------------
__global__ __launch_bounds__(256)
void attn_scores_k(const bf16_t* __restrict__ Q, const bf16_t* __restrict__ Kc,
                   const int* __restrict__ text_len, float* __restrict__ scores) {
  __shared__ bf16_t Qs[128][80];   // 64-elem rows (32 DW, code 4) + 8 DW pad (code 7)
  __shared__ bf16_t Ks[128][80];
  const int bh = blockIdx.x, b = bh / H_, h = bh % H_;
  const bf16_t* qb = Q  + (size_t)b * T_ * D_ + h * DH_;
  const bf16_t* kb = Kc + (size_t)b * T_ * D_ + h * DH_;
  const int tid = threadIdx.x;
  const int lane = tid & 31, wid = tid >> 5;
  const int wm = wid >> 2, wn = wid & 3, hl = lane >> 4, r16 = lane & 15;
  if (wid == 0) {
    tdm_load_2d(lds_off(&Qs[0][0]), qb, 64u, 128u, (unsigned long long)D_, 4u, 7u);
    tdm_load_2d(lds_off(&Ks[0][0]), kb, 64u, 128u, (unsigned long long)D_, 4u, 7u);
    __builtin_amdgcn_s_wait_tensorcnt(0);
  }
  __syncthreads();
  v8f acc[4][2];
  for (int mt = 0; mt < 4; ++mt)
    for (int nt = 0; nt < 2; ++nt)
      for (int r = 0; r < 8; ++r) acc[mt][nt][r] = 0.0f;
  for (int ks = 0; ks < 2; ++ks) {
    v16bf bfrag[2];
    for (int nt = 0; nt < 2; ++nt)
      bfrag[nt] = merge16(*(const v8bf*)&Ks[wn * 32 + nt * 16 + r16][ks * 32 + hl * 8],
                          *(const v8bf*)&Ks[wn * 32 + nt * 16 + r16][ks * 32 + 16 + hl * 8]);
    for (int mt = 0; mt < 4; ++mt) {
      const int arow = wm * 64 + mt * 16 + r16;
      v8bf alo = *(const v8bf*)&Qs[arow][ks * 32 + hl * 8];
      v8bf ahi = *(const v8bf*)&Qs[arow][ks * 32 + 16 + hl * 8];
      v16bf afrag = merge16(alo, ahi);
      for (int nt = 0; nt < 2; ++nt)
        acc[mt][nt] = wmma_bf16(afrag, bfrag[nt], acc[mt][nt]);
    }
  }
  const int len = text_len[b];
  for (int mt = 0; mt < 4; ++mt)
    for (int nt = 0; nt < 2; ++nt) {
      const int col = wn * 32 + nt * 16 + r16;
      const float mask = (col < len) ? 0.0f : -1e9f;
      for (int r = 0; r < 8; ++r) {
        const int row = wm * 64 + mt * 16 + hl * 8 + r;
        scores[((size_t)bh * T_ + row) * T_ + col] = acc[mt][nt][r] * ASCALE_ + mask;
      }
    }
}

// ---------------- softmax over keys (row of 128) ----------------
__global__ __launch_bounds__(128)
void softmax_k(const float* __restrict__ scores, bf16_t* __restrict__ P) {
  __shared__ float red[128];
  const int row = blockIdx.x, tid = threadIdx.x;
  const float v = scores[(size_t)row * T_ + tid];
  red[tid] = v; __syncthreads();
  for (int s = 64; s > 0; s >>= 1) {
    if (tid < s) red[tid] = fmaxf(red[tid], red[tid + s]);
    __syncthreads();
  }
  const float m = red[0]; __syncthreads();
  const float e = __expf(v - m);
  red[tid] = e; __syncthreads();
  for (int s = 64; s > 0; s >>= 1) {
    if (tid < s) red[tid] += red[tid + s];
    __syncthreads();
  }
  P[(size_t)row * T_ + tid] = (bf16_t)(e / red[0]);
}

// ---------------- O = P@V per (b,h): 128x64, K=128 ----------------
__global__ __launch_bounds__(256)
void attn_av_k(const bf16_t* __restrict__ P, const bf16_t* __restrict__ V,
               bf16_t* __restrict__ O) {
  __shared__ bf16_t Ps[128][144];  // 128-elem rows (64 DW, code 5) + 8 DW pad (code 7)
  __shared__ bf16_t Vs[64][144];   // [dh][t'] (transposed manually)
  const int bh = blockIdx.x, b = bh / H_, h = bh % H_;
  const int tid = threadIdx.x;
  const int lane = tid & 31, wid = tid >> 5;
  const int wm = wid >> 1, wn2 = wid & 1, hl = lane >> 4, r16 = lane & 15;
  const bf16_t* pb = P + (size_t)bh * T_ * T_;
  const bf16_t* vb = V + (size_t)b * T_ * D_ + h * DH_;
  if (wid == 0)
    tdm_load_2d(lds_off(&Ps[0][0]), pb, 128u, 128u, (unsigned long long)T_, 5u, 7u);
  for (int i = 0; i < 8; ++i) {
    const int tp = i * 16 + (tid >> 4), d0 = (tid & 15) * 4;
    v4bf vv = *(const v4bf*)(vb + (size_t)tp * D_ + d0);
    Vs[d0 + 0][tp] = vv[0]; Vs[d0 + 1][tp] = vv[1];
    Vs[d0 + 2][tp] = vv[2]; Vs[d0 + 3][tp] = vv[3];
  }
  if (wid == 0) __builtin_amdgcn_s_wait_tensorcnt(0);
  __syncthreads();
  v8f acc[2][2];
  for (int mt = 0; mt < 2; ++mt)
    for (int nt = 0; nt < 2; ++nt)
      for (int r = 0; r < 8; ++r) acc[mt][nt][r] = 0.0f;
  for (int ks = 0; ks < 4; ++ks) {
    v16bf bfrag[2];
    for (int nt = 0; nt < 2; ++nt)
      bfrag[nt] = merge16(*(const v8bf*)&Vs[wn2 * 32 + nt * 16 + r16][ks * 32 + hl * 8],
                          *(const v8bf*)&Vs[wn2 * 32 + nt * 16 + r16][ks * 32 + 16 + hl * 8]);
    for (int mt = 0; mt < 2; ++mt) {
      const int arow = wm * 32 + mt * 16 + r16;
      v8bf alo = *(const v8bf*)&Ps[arow][ks * 32 + hl * 8];
      v8bf ahi = *(const v8bf*)&Ps[arow][ks * 32 + 16 + hl * 8];
      v16bf afrag = merge16(alo, ahi);
      for (int nt = 0; nt < 2; ++nt)
        acc[mt][nt] = wmma_bf16(afrag, bfrag[nt], acc[mt][nt]);
    }
  }
  for (int mt = 0; mt < 2; ++mt)
    for (int nt = 0; nt < 2; ++nt) {
      const int col = wn2 * 32 + nt * 16 + r16;
      for (int r = 0; r < 8; ++r) {
        const int row = wm * 32 + mt * 16 + hl * 8 + r;
        O[((size_t)b * T_ + row) * D_ + h * DH_ + col] = (bf16_t)acc[mt][nt][r];
      }
    }
}

// ---------------- residual add + LayerNorm (row of 1024) ----------------
__global__ __launch_bounds__(256)
void residual_ln_k(float* __restrict__ hf, const float* __restrict__ delta,
                   const float* __restrict__ g, const float* __restrict__ bta,
                   bf16_t* __restrict__ hb) {
  __shared__ float red[256];
  const int row = blockIdx.x, tid = threadIdx.x;
  float vals[4]; float s = 0.0f;
  for (int i = 0; i < 4; ++i) {
    const int d = tid + i * 256;
    vals[i] = hf[(size_t)row * D_ + d] + delta[(size_t)row * D_ + d];
    s += vals[i];
  }
  red[tid] = s; __syncthreads();
  for (int st = 128; st > 0; st >>= 1) {
    if (tid < st) red[tid] += red[tid + st];
    __syncthreads();
  }
  const float mean = red[0] * (1.0f / D_); __syncthreads();
  float vs = 0.0f;
  for (int i = 0; i < 4; ++i) { const float dd = vals[i] - mean; vs += dd * dd; }
  red[tid] = vs; __syncthreads();
  for (int st = 128; st > 0; st >>= 1) {
    if (tid < st) red[tid] += red[tid + st];
    __syncthreads();
  }
  const float rstd = rsqrtf(red[0] * (1.0f / D_) + 1e-5f);
  for (int i = 0; i < 4; ++i) {
    const int d = tid + i * 256;
    const float y = (vals[i] - mean) * rstd * g[d] + bta[d];
    hf[(size_t)row * D_ + d] = y;
    hb[(size_t)row * D_ + d] = (bf16_t)y;
  }
}

// ---------------- host ----------------
static inline size_t alignup(size_t x) { return (x + 255) & ~(size_t)255; }

extern "C" void kernel_launch(void* const* d_in, const int* in_sizes, int n_in,
                              void* d_out, int out_size, void* d_ws, size_t ws_size,
                              hipStream_t stream) {
  const float* x           = (const float*)d_in[0];
  const int*   word_ids    = (const int*)d_in[1];
  const int*   text_len    = (const int*)d_in[2];
  const int*   pos_tags    = (const int*)d_in[3];
  const float* pos_table   = (const float*)d_in[4];
  const float* Wq = (const float*)d_in[5];   const float* bq = (const float*)d_in[6];
  const float* Wk = (const float*)d_in[7];   const float* bk = (const float*)d_in[8];
  const float* Wv = (const float*)d_in[9];   const float* bv = (const float*)d_in[10];
  const float* Wo = (const float*)d_in[11];  const float* bo = (const float*)d_in[12];
  const float* ln1_g = (const float*)d_in[13]; const float* ln1_b = (const float*)d_in[14];
  const float* W1 = (const float*)d_in[15];  const float* b1 = (const float*)d_in[16];
  const float* W2 = (const float*)d_in[17];  const float* b2 = (const float*)d_in[18];
  const float* ln2_g = (const float*)d_in[19]; const float* ln2_b = (const float*)d_in[20];
  const float* Wout = (const float*)d_in[21]; const float* bout = (const float*)d_in[22];

  char* ws = (char*)d_ws;
  size_t off = 0;
  auto take = [&](size_t bytes) { void* p = ws + off; off = alignup(off + bytes); return p; };

  float*  hf     = (float*)  take((size_t)M4_ * D_ * 4);
  bf16_t* hb     = (bf16_t*) take((size_t)M4_ * D_ * 2);
  bf16_t* qbuf   = (bf16_t*) take((size_t)M4_ * D_ * 2);
  bf16_t* kbuf   = (bf16_t*) take((size_t)M4_ * D_ * 2);
  bf16_t* vbuf   = (bf16_t*) take((size_t)M4_ * D_ * 2);
  float*  scores = (float*)  take((size_t)B_ * H_ * T_ * T_ * 4);
  bf16_t* pbuf   = (bf16_t*) take((size_t)B_ * H_ * T_ * T_ * 2);
  bf16_t* attn   = (bf16_t*) take((size_t)M4_ * D_ * 2);
  float*  delta  = (float*)  take((size_t)M4_ * D_ * 4);
  bf16_t* f1     = (bf16_t*) take((size_t)M4_ * FF_ * 2);
  // weights, transposed to N x K bf16 (TDM-friendly: rows contiguous in K)
  bf16_t* wqT    = (bf16_t*) take((size_t)D_ * D_ * 2);
  bf16_t* wkT    = (bf16_t*) take((size_t)D_ * D_ * 2);
  bf16_t* wvT    = (bf16_t*) take((size_t)D_ * D_ * 2);
  bf16_t* woT    = (bf16_t*) take((size_t)D_ * D_ * 2);
  bf16_t* w1T    = (bf16_t*) take((size_t)FF_ * D_ * 2);    // 4096 x 1024
  bf16_t* w2T    = (bf16_t*) take((size_t)D_ * FF_ * 2);    // 1024 x 4096
  bf16_t* woutT  = (bf16_t*) take((size_t)NSENSEP * D_ * 2); // 5120 x 1024, pad rows 0

  auto cvtT = [&](const float* s, bf16_t* d, int K, int N, int Ndst) {
    transpose_to_bf16_k<<<dim3(Ndst / 32, K / 32), 256, 0, stream>>>(s, d, K, N);
  };
  cvtT(Wq, wqT, D_, D_, D_);     cvtT(Wk, wkT, D_, D_, D_);
  cvtT(Wv, wvT, D_, D_, D_);     cvtT(Wo, woT, D_, D_, D_);
  cvtT(W1, w1T, D_, FF_, FF_);   cvtT(W2, w2T, FF_, D_, D_);
  cvtT(Wout, woutT, D_, NSENSE_, NSENSEP);

  pool_embed_k<<<M4_, 256, 0, stream>>>(x, word_ids, pos_tags, pos_table, hf, hb);

  const dim3 gD(D_ / 128, M4_ / 128);        // (8,32)
  const dim3 gFF(FF_ / 128, M4_ / 128);      // (32,32)
  const dim3 gOut(NSENSEP / 128, M4_ / 128); // (40,32)

  for (int l = 0; l < 2; ++l) {
    gemm_wmma_k<0><<<gD, 256, 0, stream>>>(hb, wqT, bq, qbuf, D_, D_);
    gemm_wmma_k<0><<<gD, 256, 0, stream>>>(hb, wkT, bk, kbuf, D_, D_);
    gemm_wmma_k<0><<<gD, 256, 0, stream>>>(hb, wvT, bv, vbuf, D_, D_);
    attn_scores_k<<<B_ * H_, 256, 0, stream>>>(qbuf, kbuf, text_len, scores);
    softmax_k<<<B_ * H_ * T_, 128, 0, stream>>>(scores, pbuf);
    attn_av_k<<<B_ * H_, 256, 0, stream>>>(pbuf, vbuf, attn);
    gemm_wmma_k<2><<<gD, 256, 0, stream>>>(attn, woT, bo, delta, D_, D_);
    residual_ln_k<<<M4_, 256, 0, stream>>>(hf, delta, ln1_g, ln1_b, hb);
    gemm_wmma_k<1><<<gFF, 256, 0, stream>>>(hb, w1T, b1, f1, FF_, D_);
    gemm_wmma_k<2><<<gD, 256, 0, stream>>>(f1, w2T, b2, delta, D_, FF_);
    residual_ln_k<<<M4_, 256, 0, stream>>>(hf, delta, ln2_g, ln2_b, hb);
  }
  gemm_wmma_k<2><<<gOut, 256, 0, stream>>>(hb, woutT, bout, (float*)d_out,
                                           NSENSE_, D_);
}
